// _LocalState_76991583748295
// MI455X (gfx1250) — matrix operand order repaired and checked by reading, not compiled, not run on hardware
//
#include <hip/hip_runtime.h>
#include <hip/hip_bf16.h>

#define BB     4
#define CC     1024
#define TT     2048
#define HEADSN 8
#define HDIM   128
#define NDEC   4

typedef __bf16 bf16_t;
typedef __attribute__((ext_vector_type(16))) __bf16 v16bf;
typedef __attribute__((ext_vector_type(8)))  __bf16 v8bf;
typedef __attribute__((ext_vector_type(8)))  float  v8f;

// Native f32->bf16 convert: lowers to v_cvt_pk_bf16_f32 on gfx1250 instead of
// a 4-op integer round-to-nearest-even sequence.
static __device__ __forceinline__ bf16_t f2bf(float f) {
  return (bf16_t)f;
}

// ---------------------------------------------------------------------------
// x [B,C,T] f32  ->  xT [B,T,C] bf16   (LDS tile transpose, coalesced)
// ---------------------------------------------------------------------------
__global__ void k_xT(const float* __restrict__ x, bf16_t* __restrict__ xT) {
  __shared__ float tile[16][17];
  const int b  = blockIdx.z;
  const int c0 = blockIdx.y * 16;
  const int t0 = blockIdx.x * 16;
  const int tx = threadIdx.x, ty = threadIdx.y;
  tile[ty][tx] = x[(size_t)b * CC * TT + (size_t)(c0 + ty) * TT + (t0 + tx)];
  __syncthreads();
  xT[(size_t)b * TT * CC + (size_t)(t0 + ty) * CC + (c0 + tx)] = f2bf(tile[tx][ty]);
}

__global__ void k_w2bf(const float* __restrict__ w, bf16_t* __restrict__ o, int n) {
  int i = blockIdx.x * 256 + threadIdx.x;
  if (i < n) o[i] = f2bf(w[i]);
}

// ---------------------------------------------------------------------------
// GEMM: out = W (CoutxC) @ x_b (CxT) + bias, via v_wmma_f32_16x16x32_bf16.
// 2x4 register-blocked: one wave computes a 32(o) x 64(t) block (8 D tiles),
// so each A operand feeds 4 WMMAs and each B operand feeds 2 (24 B/lane/WMMA).
// Xt is [B,T,C] bf16 so all operands are contiguous 16B loads, A and B in the same
// contiguous k order (lane half h holds k = 16h..16h+15).
// mode 0: store bf16 [B,H,T,HD] (q,k)   mode 1: store bf16 [B,C,T] (content)
// mode 2: f32 [B,C,T] = residual + bias + acc (final proj)
// ---------------------------------------------------------------------------
__global__ void __launch_bounds__(32)
k_gemm(const bf16_t* __restrict__ W, const bf16_t* __restrict__ Xt,
       const float* __restrict__ bias, void* __restrict__ out,
       const float* __restrict__ xres, int mode) {
  const int lane = threadIdx.x;
  const bool hi  = lane >= 16;
  const int  l   = lane & 15;
  const int  b   = blockIdx.z;
  const int  o0  = blockIdx.y * 32;
  const int  t0  = blockIdx.x * 64;

  const bf16_t* ap0 = W + (size_t)(o0 + l) * CC + (hi ? 16 : 0);
  const bf16_t* ap1 = ap0 + (size_t)16 * CC;
  const bf16_t* xb  = Xt + (size_t)b * TT * CC + (hi ? 16 : 0);
  const bf16_t* bp0 = xb + (size_t)(t0 + l) * CC;
  const bf16_t* bp1 = bp0 + (size_t)16 * CC;
  const bf16_t* bp2 = bp0 + (size_t)32 * CC;
  const bf16_t* bp3 = bp0 + (size_t)48 * CC;

  v8f acc[2][4];
#pragma unroll
  for (int oi = 0; oi < 2; ++oi)
#pragma unroll
    for (int tj = 0; tj < 4; ++tj) acc[oi][tj] = (v8f){};

#pragma unroll 1
  for (int c0 = 0; c0 < CC; c0 += 32) {
    v16bf Aop[2], Bop[4];
    {
      v8bf lo0 = *(const v8bf*)(ap0 + c0);
      v8bf hi0 = *(const v8bf*)(ap0 + c0 + 8);
      v8bf lo1 = *(const v8bf*)(ap1 + c0);
      v8bf hi1 = *(const v8bf*)(ap1 + c0 + 8);
#pragma unroll
      for (int i = 0; i < 8; ++i) {
        Aop[0][i] = lo0[i]; Aop[0][8 + i] = hi0[i];
        Aop[1][i] = lo1[i]; Aop[1][8 + i] = hi1[i];
      }
    }
    {
      v8bf l0 = *(const v8bf*)(bp0 + c0); v8bf h0 = *(const v8bf*)(bp0 + c0 + 8);
      v8bf l1 = *(const v8bf*)(bp1 + c0); v8bf h1 = *(const v8bf*)(bp1 + c0 + 8);
      v8bf l2 = *(const v8bf*)(bp2 + c0); v8bf h2 = *(const v8bf*)(bp2 + c0 + 8);
      v8bf l3 = *(const v8bf*)(bp3 + c0); v8bf h3 = *(const v8bf*)(bp3 + c0 + 8);
#pragma unroll
      for (int i = 0; i < 8; ++i) {
        Bop[0][i] = l0[i]; Bop[0][8 + i] = h0[i];
        Bop[1][i] = l1[i]; Bop[1][8 + i] = h1[i];
        Bop[2][i] = l2[i]; Bop[2][8 + i] = h2[i];
        Bop[3][i] = l3[i]; Bop[3][8 + i] = h3[i];
      }
    }
#pragma unroll
    for (int oi = 0; oi < 2; ++oi)
#pragma unroll
      for (int tj = 0; tj < 4; ++tj)
        acc[oi][tj] = __builtin_amdgcn_wmma_f32_16x16x32_bf16(
            false, Aop[oi], false, Bop[tj], (short)0, acc[oi][tj], false, false);
  }

  const int r0 = hi ? 8 : 0;
#pragma unroll
  for (int oi = 0; oi < 2; ++oi) {
    const int oo = o0 + oi * 16;
    float bias8[8];
#pragma unroll
    for (int i = 0; i < 8; ++i) bias8[i] = bias[oo + r0 + i];
#pragma unroll
    for (int tj = 0; tj < 4; ++tj) {
      const int tt = t0 + tj * 16;
      v8f a = acc[oi][tj];
      if (mode == 0) {                   // [B,H,T,HD] bf16
        const int h   = oo / HDIM;
        const int hd0 = (oo % HDIM) + r0;
        bf16_t* dst = (bf16_t*)out +
            (((size_t)b * HEADSN + h) * TT + (tt + l)) * HDIM + hd0;
        v8bf v;
#pragma unroll
        for (int i = 0; i < 8; ++i) v[i] = f2bf(a[i] + bias8[i]);
        *(v8bf*)dst = v;
      } else if (mode == 1) {            // [B,C,T] bf16
        bf16_t* dst = (bf16_t*)out + ((size_t)b * CC + oo + r0) * TT + (tt + l);
#pragma unroll
        for (int i = 0; i < 8; ++i) dst[(size_t)i * TT] = f2bf(a[i] + bias8[i]);
      } else {                           // f32 residual output
        const size_t base = ((size_t)b * CC + oo + r0) * TT + (tt + l);
        float* dp = (float*)out + base;
        const float* xr = xres + base;
#pragma unroll
        for (int i = 0; i < 8; ++i)
          dp[(size_t)i * TT] = xr[(size_t)i * TT] + a[i] + bias8[i];
      }
    }
  }
}

// ---------------------------------------------------------------------------
// coef[b,h,s] = (1/sqrt(NDEC)) * sum_f (f+1) * sigmoid(Wd[h,f,:]·x[:,s]-2)/2
// ---------------------------------------------------------------------------
__global__ void k_decay(const float* __restrict__ x, const float* __restrict__ Wd,
                        const float* __restrict__ bd, float* __restrict__ coef) {
  const int s = blockIdx.x * blockDim.x + threadIdx.x;
  const int h = blockIdx.y, b = blockIdx.z;
  float acc[NDEC];
#pragma unroll
  for (int f = 0; f < NDEC; ++f) acc[f] = bd[h * NDEC + f];
  const float* xb = x + (size_t)b * CC * TT + s;
  for (int c = 0; c < CC; ++c) {
    float xv = xb[(size_t)c * TT];
#pragma unroll
    for (int f = 0; f < NDEC; ++f) acc[f] += Wd[(h * NDEC + f) * CC + c] * xv;
  }
  float cf = 0.0f;
#pragma unroll
  for (int f = 0; f < NDEC; ++f) {
    float sg = 1.0f / (1.0f + __expf(-acc[f]));
    cf += (float)(f + 1) * (0.5f * sg);
  }
  coef[((size_t)b * HEADSN + h) * TT + s] = 0.5f * cf;   // /sqrt(4)
}

// ---------------------------------------------------------------------------
// Fused flash attention: one wave per (b, h, 16 s-columns).
// dots[t,s] = K^T Q / sqrt(HD) - |t-s|*coef[s];  diag -> -100;
// online softmax over t; O[c,s] = content @ P;  store [B,T,C] bf16.
// ---------------------------------------------------------------------------
__global__ void __launch_bounds__(32)
k_attn(const bf16_t* __restrict__ Q, const bf16_t* __restrict__ K,
       const bf16_t* __restrict__ V, const float* __restrict__ coef,
       bf16_t* __restrict__ Ot) {
  const int lane = threadIdx.x;
  const bool hi  = lane >= 16;
  const int  l   = lane & 15;
  const int  b   = blockIdx.z, h = blockIdx.y;
  const int  s0  = blockIdx.x * 16;
  const int  scol = s0 + l;
  const int  r0   = hi ? 8 : 0;

  const bf16_t* qb = Q + ((size_t)b * HEADSN + h) * TT * HDIM;
  const bf16_t* kb = K + ((size_t)b * HEADSN + h) * TT * HDIM;
  const bf16_t* vb = V + ((size_t)b * CC + h * HDIM) * TT;
  const float   cf = coef[((size_t)b * HEADSN + h) * TT + scol];
  const float scale = 0.08838834764831845f;   // 1/sqrt(128)

  // Q operands: loop-invariant, keep in registers (4 c-steps of K=32)
  v16bf Qop[4];
  const bf16_t* qrow = qb + (size_t)scol * HDIM + (hi ? 8 : 0);
#pragma unroll
  for (int cs = 0; cs < 4; ++cs) {
    v8bf lo = *(const v8bf*)(qrow + cs * 32);
    v8bf hh = *(const v8bf*)(qrow + cs * 32 + 16);
#pragma unroll
    for (int i = 0; i < 8; ++i) { Qop[cs][i] = lo[i]; Qop[cs][8 + i] = hh[i]; }
  }

  v8f O[8];
#pragma unroll
  for (int j = 0; j < 8; ++j) O[j] = (v8f){};
  float mrun = -1e30f, lrun = 0.0f;

#pragma unroll 1
  for (int t0 = 0; t0 < TT; t0 += 32) {
    const bf16_t* krow0 = kb + (size_t)(t0 + l) * HDIM + (hi ? 8 : 0);
    const bf16_t* krow1 = krow0 + 16 * HDIM;
    __builtin_prefetch((const void*)(krow0 + 32 * HDIM), 0, 0);

    v8f S0 = {}, S1 = {};
#pragma unroll
    for (int cs = 0; cs < 4; ++cs) {
      v8bf a0 = *(const v8bf*)(krow0 + cs * 32);
      v8bf a1 = *(const v8bf*)(krow0 + cs * 32 + 16);
      v16bf A;
#pragma unroll
      for (int i = 0; i < 8; ++i) { A[i] = a0[i]; A[8 + i] = a1[i]; }
      S0 = __builtin_amdgcn_wmma_f32_16x16x32_bf16(false, A, false, Qop[cs],
                                                   (short)0, S0, false, false);
      v8bf c0v = *(const v8bf*)(krow1 + cs * 32);
      v8bf c1v = *(const v8bf*)(krow1 + cs * 32 + 16);
      v16bf A1;
#pragma unroll
      for (int i = 0; i < 8; ++i) { A1[i] = c0v[i]; A1[8 + i] = c1v[i]; }
      S1 = __builtin_amdgcn_wmma_f32_16x16x32_bf16(false, A1, false, Qop[cs],
                                                   (short)0, S1, false, false);
    }

    // bias + mask + online softmax (per column s; stats replicated lane/lane+16)
    float e0[8], e1[8];
    float mloc = -1e30f;
#pragma unroll
    for (int i = 0; i < 8; ++i) {
      int row0 = t0 + r0 + i;
      int row1 = row0 + 16;
      float d0 = S0[i] * scale - fabsf((float)(row0 - scol)) * cf;
      float d1 = S1[i] * scale - fabsf((float)(row1 - scol)) * cf;
      if (row0 == scol) d0 = -100.0f;
      if (row1 == scol) d1 = -100.0f;
      e0[i] = d0; e1[i] = d1;
      mloc = fmaxf(mloc, fmaxf(d0, d1));
    }
    mloc = fmaxf(mloc, __shfl_xor(mloc, 16, 32));
    float mnew = fmaxf(mrun, mloc);
    float corr = __expf(mrun - mnew);
    float psum = 0.0f;
#pragma unroll
    for (int i = 0; i < 8; ++i) {
      e0[i] = __expf(e0[i] - mnew);
      e1[i] = __expf(e1[i] - mnew);
      psum += e0[i] + e1[i];
    }
    psum += __shfl_xor(psum, 16, 32);
    lrun = lrun * corr + psum;
    mrun = mnew;
#pragma unroll
    for (int j = 0; j < 8; ++j)
#pragma unroll
      for (int i = 0; i < 8; ++i) O[j][i] *= corr;

    // P D-tiles -> 32x16 bf16 B operand (lane<->lane+16 exchange)
    float p0sw[8], p1sw[8];
#pragma unroll
    for (int i = 0; i < 8; ++i) {
      p0sw[i] = __shfl_xor(e0[i], 16, 32);
      p1sw[i] = __shfl_xor(e1[i], 16, 32);
    }
    v16bf Pop;
#pragma unroll
    for (int i = 0; i < 8; ++i) {
      Pop[i]     = f2bf(hi ? p1sw[i] : e0[i]);   // K = 0..7  (or 16..23)
      Pop[8 + i] = f2bf(hi ? e1[i]   : p0sw[i]); // K = 8..15 (or 24..31)
    }

    // O[c-tile] += content(A) x P(B)
#pragma unroll
    for (int j = 0; j < 8; ++j) {
      const bf16_t* ar = vb + (size_t)(j * 16 + l) * TT + t0 + (hi ? 16 : 0);
      v8bf a0 = *(const v8bf*)(ar);
      v8bf a1 = *(const v8bf*)(ar + 8);
      v16bf A;
#pragma unroll
      for (int i = 0; i < 8; ++i) { A[i] = a0[i]; A[8 + i] = a1[i]; }
      O[j] = __builtin_amdgcn_wmma_f32_16x16x32_bf16(false, A, false, Pop,
                                                     (short)0, O[j], false, false);
    }
  }

  const float inv = 1.0f / lrun;
  bf16_t* orow = Ot + ((size_t)b * TT + scol) * CC + h * HDIM;
#pragma unroll
  for (int j = 0; j < 8; ++j) {
    v8bf v;
#pragma unroll
    for (int i = 0; i < 8; ++i) v[i] = f2bf(O[j][i] * inv);
    *(v8bf*)(orow + j * 16 + r0) = v;
  }
}

// ---------------------------------------------------------------------------
extern "C" void kernel_launch(void* const* d_in, const int* in_sizes, int n_in,
                              void* d_out, int out_size, void* d_ws, size_t ws_size,
                              hipStream_t stream) {
  const float* x         = (const float*)d_in[0];
  const float* W_content = (const float*)d_in[1];
  const float* b_content = (const float*)d_in[2];
  const float* W_query   = (const float*)d_in[3];
  const float* b_query   = (const float*)d_in[4];
  const float* W_key     = (const float*)d_in[5];
  const float* b_key     = (const float*)d_in[6];
  const float* W_qdecay  = (const float*)d_in[7];
  const float* b_qdecay  = (const float*)d_in[8];
  const float* W_proj    = (const float*)d_in[9];
  const float* b_proj    = (const float*)d_in[10];

  char* ws = (char*)d_ws;
  size_t off = 0;
  auto take = [&](size_t bytes) -> char* {
    char* p = ws + off;
    off += (bytes + 255) & ~(size_t)255;
    return p;
  };
  bf16_t* xT   = (bf16_t*)take((size_t)BB * TT * CC * 2);
  bf16_t* Wq   = (bf16_t*)take((size_t)CC * CC * 2);
  bf16_t* Wk   = (bf16_t*)take((size_t)CC * CC * 2);
  bf16_t* Wc   = (bf16_t*)take((size_t)CC * CC * 2);
  bf16_t* Wp   = (bf16_t*)take((size_t)CC * CC * 2);
  bf16_t* qB   = (bf16_t*)take((size_t)BB * HEADSN * TT * HDIM * 2);
  bf16_t* kB   = (bf16_t*)take((size_t)BB * HEADSN * TT * HDIM * 2);
  bf16_t* cont = (bf16_t*)take((size_t)BB * CC * TT * 2);
  bf16_t* attn = (bf16_t*)take((size_t)BB * TT * CC * 2);
  float*  coef = (float*)take((size_t)BB * HEADSN * TT * 4);
  (void)in_sizes; (void)n_in; (void)out_size; (void)ws_size;

  k_xT<<<dim3(TT / 16, CC / 16, BB), dim3(16, 16, 1), 0, stream>>>(x, xT);

  const int nW = CC * CC;
  k_w2bf<<<dim3(nW / 256), 256, 0, stream>>>(W_query, Wq, nW);
  k_w2bf<<<dim3(nW / 256), 256, 0, stream>>>(W_key, Wk, nW);
  k_w2bf<<<dim3(nW / 256), 256, 0, stream>>>(W_content, Wc, nW);
  k_w2bf<<<dim3(nW / 256), 256, 0, stream>>>(W_proj, Wp, nW);

  dim3 gg(TT / 64, CC / 32, BB);
  k_gemm<<<gg, 32, 0, stream>>>(Wq, xT, b_query, (void*)qB, nullptr, 0);
  k_gemm<<<gg, 32, 0, stream>>>(Wk, xT, b_key, (void*)kB, nullptr, 0);
  k_gemm<<<gg, 32, 0, stream>>>(Wc, xT, b_content, (void*)cont, nullptr, 1);

  k_decay<<<dim3(TT / 256, HEADSN, BB), 256, 0, stream>>>(x, W_qdecay, b_qdecay, coef);

  k_attn<<<dim3(TT / 16, HEADSN, BB), 32, 0, stream>>>(qB, kB, cont, coef, attn);

  k_gemm<<<gg, 32, 0, stream>>>(Wp, attn, b_proj, d_out, x, 2);
}
